// InfoNCE_46042049413624
// MI455X (gfx1250) — matrix operand order repaired and checked by reading, not compiled, not run on hardware
//
#include <hip/hip_runtime.h>
#include <math.h>

// ---------------------------------------------------------------------------
// InfoNCE (NT-Xent) fused kernel for MI455X / gfx1250, wave32 + WMMA f16.
//   B = 4096 rows per view, D = 256, N2 = 8192 total rows, T = 0.07
// Phase 1: L2-normalize rows of [data1;data2] -> f16 z in d_ws (4 MB)
// Phase 2: fused z*z^T. Blocks of 4 waves; each wave owns TWO 16-row A tiles
//          (each LDS B fragment feeds 2 WMMAs). 64-column B panels are
//          staged into double-buffered LDS via GLOBAL_LOAD_ASYNC_TO_LDS
//          (ASYNCcnt); B fragments are software-pipelined 1-deep through
//          two register buffers so DS waits are partial, not full.
//          Online exp-sum with fixed shift m=1 (cosines <= 1), diagonal
//          mask, pos on the +/-B diagonal, atomic mean into d_out.
// ---------------------------------------------------------------------------

typedef _Float16 v16h __attribute__((ext_vector_type(16)));
typedef _Float16 v8h  __attribute__((ext_vector_type(8)));
typedef float    v8f  __attribute__((ext_vector_type(8)));
typedef float    v4f  __attribute__((ext_vector_type(4)));
typedef int      v4i  __attribute__((vector_size(16)));   // matches builtin sig

#define BROWS 4096
#define N2    8192
#define DDIM  256
#define INV_T (1.0f / 0.07f)

// LDS staging: 64 columns (rows of z) per round, double buffered.
#define CROWS   64                    // columns staged per round
#define ROUNDS  (N2 / CROWS)          // 128
#define LSTRIDE 264                   // padded row stride in halves
                                      // (132 words == 4 mod 64 banks)

#if __has_builtin(__builtin_amdgcn_global_load_async_to_lds_b128)
#define HAVE_ASYNC_LDS 1
#else
#define HAVE_ASYNC_LDS 0
#endif

#if HAVE_ASYNC_LDS
__device__ __forceinline__ void async_b128(const _Float16* g, _Float16* l) {
    // (v4i __device__*, v4i __shared__*, imm offset, imm cpol);
    // __device__ == addrspace(1), __shared__ == addrspace(3).
    __builtin_amdgcn_global_load_async_to_lds_b128(
        (__attribute__((address_space(1))) v4i*)(v4i*)g,
        (__attribute__((address_space(3))) v4i*)(v4i*)l,
        /*offset=*/0, /*cpol=*/0);
}
__device__ __forceinline__ void wait_async0(void) {
#if __has_builtin(__builtin_amdgcn_s_wait_asynccnt)
    __builtin_amdgcn_s_wait_asynccnt(0);
#else
    asm volatile("s_wait_asynccnt 0x0" ::: "memory");
#endif
}
#endif

__global__ void infonce_zero_out(float* out) {
    if (threadIdx.x == 0 && blockIdx.x == 0) out[0] = 0.0f;
}

// One wave (32 lanes) per row; each lane handles 8 contiguous floats.
__global__ __launch_bounds__(256) void infonce_normalize_f16(
    const float* __restrict__ d1, const float* __restrict__ d2,
    _Float16* __restrict__ z)
{
    const int wave = threadIdx.x >> 5;
    const int lane = threadIdx.x & 31;
    const int row  = blockIdx.x * 8 + wave;

    const float* src = (row < BROWS) ? (d1 + (size_t)row * DDIM)
                                     : (d2 + (size_t)(row - BROWS) * DDIM);
    v4f x0 = *(const v4f*)(src + lane * 8);
    v4f x1 = *(const v4f*)(src + lane * 8 + 4);

    float ss = x0[0]*x0[0] + x0[1]*x0[1] + x0[2]*x0[2] + x0[3]*x0[3]
             + x1[0]*x1[0] + x1[1]*x1[1] + x1[2]*x1[2] + x1[3]*x1[3];
#pragma unroll
    for (int m = 1; m < 32; m <<= 1) ss += __shfl_xor(ss, m, 32);

    const float inv = 1.0f / fmaxf(sqrtf(ss), 1e-8f);

    v8h o;
    o[0] = (_Float16)(x0[0] * inv); o[1] = (_Float16)(x0[1] * inv);
    o[2] = (_Float16)(x0[2] * inv); o[3] = (_Float16)(x0[3] * inv);
    o[4] = (_Float16)(x1[0] * inv); o[5] = (_Float16)(x1[1] * inv);
    o[6] = (_Float16)(x1[2] * inv); o[7] = (_Float16)(x1[3] * inv);
    *(v8h*)(z + (size_t)row * DDIM + lane * 8) = o;
}

// Load one B fragment (two 8-half K runs per lane, at +0 and +16) from LDS.
__device__ __forceinline__ v16h load_bfrag(const _Float16* bp, int kc) {
    v8h lo = *(const v8h*)(bp + kc * 32);
    v8h hi = *(const v8h*)(bp + kc * 32 + 16);
    return __builtin_shufflevector(lo, hi,
            0,1,2,3,4,5,6,7,8,9,10,11,12,13,14,15);
}

// 4 waves/block (128 threads); each wave owns rows [rb0, rb0+32).
__global__ __launch_bounds__(128) void infonce_wmma_lse(
    const _Float16* __restrict__ z, float* __restrict__ out)
{
    __shared__ _Float16 smem[2][CROWS * LSTRIDE];   // 2 x 33 KB

    const int lane = threadIdx.x & 31;
    const int r16  = lane & 15;       // A: row-in-tile M; B/C: col-in-tile N
    const int hsel = lane >> 4;       // lane half (K sub-block / M+8)

    // Wave-uniform values -> SGPRs so diag/pos branches stay scalar and
    // EXEC remains all-ones around the WMMAs.
    const int rb0 = __builtin_amdgcn_readfirstlane(
        (blockIdx.x * 4 + (threadIdx.x >> 5)) * 32);
    const int rb1 = rb0 + 16;
    const int pb0 = __builtin_amdgcn_readfirstlane((rb0 + BROWS) & (N2 - 1));
    const int pb1 = pb0 + 16;
    const int Mof = 8 * hsel;         // C-layout: M = v + 8*hsel, N = r16

    // Copy-in assignment: 2 threads per staged row, 128 halves each.
    const int crow = threadIdx.x >> 1;              // 0..63
    const int cq   = (threadIdx.x & 1) * 128;       // 0 or 128 halves

    // --- A fragments: 16x32 f16, interleaved K layout (ISA 7.12.2) ---------
    // halves 0..7  = contiguous K at kc*32 + hsel*8
    // halves 8..15 = contiguous K at kc*32 + 16 + hsel*8
    v16h a0[8], a1[8];
    {
        const _Float16* ar0 = z + (size_t)(rb0 + r16) * DDIM;
        const _Float16* ar1 = z + (size_t)(rb1 + r16) * DDIM;
#pragma unroll
        for (int kc = 0; kc < 8; ++kc) {
            v8h lo = *(const v8h*)(ar0 + kc * 32 + hsel * 8);
            v8h hi = *(const v8h*)(ar0 + kc * 32 + 16 + hsel * 8);
            a0[kc] = __builtin_shufflevector(lo, hi,
                      0,1,2,3,4,5,6,7,8,9,10,11,12,13,14,15);
        }
#pragma unroll
        for (int kc = 0; kc < 8; ++kc) {
            v8h lo = *(const v8h*)(ar1 + kc * 32 + hsel * 8);
            v8h hi = *(const v8h*)(ar1 + kc * 32 + 16 + hsel * 8);
            a1[kc] = __builtin_shufflevector(lo, hi,
                      0,1,2,3,4,5,6,7,8,9,10,11,12,13,14,15);
        }
    }

    float sums0[8], pos0[8], sums1[8], pos1[8];
#pragma unroll
    for (int v = 0; v < 8; ++v) {
        sums0[v] = 0.0f; pos0[v] = 0.0f;
        sums1[v] = 0.0f; pos1[v] = 0.0f;
    }

    // Prologue: stage round 0 into buffer 0.
#if HAVE_ASYNC_LDS
    {
        const _Float16* srcp = z + (size_t)crow * DDIM + cq;
        _Float16* dstp = &smem[0][crow * LSTRIDE + cq];
#pragma unroll
        for (int i = 0; i < 16; ++i)
            async_b128(srcp + i * 8, dstp + i * 8);
        wait_async0();
    }
    __syncthreads();
#else
    {
        const _Float16* srcp = z + (size_t)crow * DDIM + cq;
        _Float16* dstp = &smem[0][crow * LSTRIDE + cq];
#pragma unroll
        for (int i = 0; i < 16; ++i)
            *(v8h*)(dstp + i * 8) = *(const v8h*)(srcp + i * 8);
    }
    __syncthreads();
#endif

    for (int round = 0; round < ROUNDS; ++round) {
        const int p = round & 1;

#if HAVE_ASYNC_LDS
        // Kick next round's panel straight into LDS (buffer p^1, last read
        // in round-1 and fenced by that round's barrier). Completion is
        // awaited with s_wait_asynccnt just before this round's barrier,
        // so the DMA runs behind the 64 WMMAs below. No VGPRs consumed.
        {
            const _Float16* srcp = z
                + (size_t)((round + 1) & (ROUNDS - 1)) * (CROWS * DDIM)
                + (size_t)crow * DDIM + cq;
            _Float16* dstp = &smem[p ^ 1][crow * LSTRIDE + cq];
            __builtin_prefetch(z
                + (size_t)((round + 3) & (ROUNDS - 1)) * (CROWS * DDIM)
                + (size_t)crow * DDIM + cq, 0, 1);
#pragma unroll
            for (int i = 0; i < 16; ++i)
                async_b128(srcp + i * 8, dstp + i * 8);
        }
#else
        v8h creg[16];
        {
            const _Float16* srcp = z
                + (size_t)((round + 1) & (ROUNDS - 1)) * (CROWS * DDIM)
                + (size_t)crow * DDIM + cq;
            __builtin_prefetch(z
                + (size_t)((round + 3) & (ROUNDS - 1)) * (CROWS * DDIM)
                + (size_t)crow * DDIM + cq, 0, 1);
#pragma unroll
            for (int i = 0; i < 16; ++i)
                creg[i] = *(const v8h*)(srcp + i * 8);
        }
#endif

        // Compute 4 column tiles from the current LDS buffer; each B
        // fragment feeds the two A row tiles (2 WMMAs per fragment).
        // Fragments are software-pipelined 1-deep (bcur/bnext), including
        // across tile boundaries and over the tile epilogues, so the DS
        // waits are partial (older pair only) instead of full drains.
        const _Float16* bbase = &smem[p][r16 * LSTRIDE + hsel * 8];
        v16h bcur = load_bfrag(bbase, 0);   // tile 0, fragment 0

#pragma unroll
        for (int ctl = 0; ctl < 4; ++ctl) {
            const int colBase = round * CROWS + ctl * 16;
            const _Float16* bp  = bbase + (ctl * 16) * LSTRIDE;
            const _Float16* bpn = bp + 16 * LSTRIDE;   // next tile's rows

            v8f c0 = {0.0f, 0.0f, 0.0f, 0.0f, 0.0f, 0.0f, 0.0f, 0.0f};
            v8f c1 = {0.0f, 0.0f, 0.0f, 0.0f, 0.0f, 0.0f, 0.0f, 0.0f};
#pragma unroll
            for (int kc = 0; kc < 8; ++kc) {
                v16h bnext = bcur;
                if (kc < 7)
                    bnext = load_bfrag(bp, kc + 1);     // 1-deep lookahead
                else if (ctl < 3)
                    bnext = load_bfrag(bpn, 0);         // cross-tile preload
                c0 = __builtin_amdgcn_wmma_f32_16x16x32_f16(
                         false, a0[kc], false, bcur, (short)0, c0,
                         false, false);
                c1 = __builtin_amdgcn_wmma_f32_16x16x32_f16(
                         false, a1[kc], false, bcur, (short)0, c1,
                         false, false);
                bcur = bnext;
            }

            // ---- epilogue, row tile 0 (scalar compares) ----
            if (colBase == rb0) {         // exclude main diagonal (N == M)
#pragma unroll
                for (int v = 0; v < 8; ++v) {
                    float e = __expf((c0[v] - 1.0f) * INV_T);
                    sums0[v] += (r16 == v + Mof) ? 0.0f : e;
                }
            } else if (colBase == pb0) {  // +/-B diagonal: in lse AND is pos
#pragma unroll
                for (int v = 0; v < 8; ++v) {
                    float s = c0[v];
                    sums0[v] += __expf((s - 1.0f) * INV_T);
                    pos0[v] = (r16 == v + Mof) ? s : pos0[v];
                }
            } else {
#pragma unroll
                for (int v = 0; v < 8; ++v)
                    sums0[v] += __expf((c0[v] - 1.0f) * INV_T);
            }

            // ---- epilogue, row tile 1 ----
            if (colBase == rb1) {
#pragma unroll
                for (int v = 0; v < 8; ++v) {
                    float e = __expf((c1[v] - 1.0f) * INV_T);
                    sums1[v] += (r16 == v + Mof) ? 0.0f : e;
                }
            } else if (colBase == pb1) {
#pragma unroll
                for (int v = 0; v < 8; ++v) {
                    float s = c1[v];
                    sums1[v] += __expf((s - 1.0f) * INV_T);
                    pos1[v] = (r16 == v + Mof) ? s : pos1[v];
                }
            } else {
#pragma unroll
                for (int v = 0; v < 8; ++v)
                    sums1[v] += __expf((c1[v] - 1.0f) * INV_T);
            }
        }

#if HAVE_ASYNC_LDS
        wait_async0();          // next panel fully landed in LDS
        __syncthreads();        // one barrier per round
#else
        __syncthreads();
        {
            _Float16* dstp = &smem[p ^ 1][crow * LSTRIDE + cq];
#pragma unroll
            for (int i = 0; i < 16; ++i)
                *(v8h*)(dstp + i * 8) = creg[i];
        }
        __syncthreads();
#endif
    }

    // Reduce across the 16 lanes that share each row M; lanes 0..15 hold
    // rows v, lanes 16..31 rows v+8 (xor masks 1..8 stay within halves).
#pragma unroll
    for (int v = 0; v < 8; ++v) {
        float s0 = sums0[v], p0 = pos0[v];
        float s1 = sums1[v], p1 = pos1[v];
#pragma unroll
        for (int m = 1; m < 16; m <<= 1) {
            s0 += __shfl_xor(s0, m, 32);
            p0 += __shfl_xor(p0, m, 32);
            s1 += __shfl_xor(s1, m, 32);
            p1 += __shfl_xor(p1, m, 32);
        }
        if (r16 == 0) {
            // lse = 1/T + log(sum_exp((s-1)/T));  term = lse - pos/T
            float t0 = INV_T + __logf(s0) - p0 * INV_T;
            float t1 = INV_T + __logf(s1) - p1 * INV_T;
            atomicAdd(out, (t0 + t1) * (1.0f / (float)N2));
        }
    }
}

extern "C" void kernel_launch(void* const* d_in, const int* in_sizes, int n_in,
                              void* d_out, int out_size, void* d_ws, size_t ws_size,
                              hipStream_t stream) {
    const float* d1 = (const float*)d_in[0];   // [4096, 256] f32
    const float* d2 = (const float*)d_in[1];   // [4096, 256] f32
    float* out = (float*)d_out;                // scalar loss
    _Float16* z = (_Float16*)d_ws;             // [8192, 256] f16 (4 MB)

    infonce_zero_out<<<1, 32, 0, stream>>>(out);
    infonce_normalize_f16<<<N2 / 8, 256, 0, stream>>>(d1, d2, z);
    // 8192 rows / (32 rows per wave * 4 waves per block) = 64 blocks
    infonce_wmma_lse<<<N2 / (32 * 4), 128, 0, stream>>>(z, out);
}
